// GNNEncoder_41351945125988
// MI455X (gfx1250) — matrix operand order repaired and checked by reading, not compiled, not run on hardware
//
#include <hip/hip_runtime.h>

// ---------------------------------------------------------------------------
// 2-layer GCN for MI455X (gfx1250, wave32, WMMA).
//  - GEMMs: v_wmma_f32_16x16x32_bf16, weights staged to LDS via
//    global_load_async_to_lds_b128 (ASYNCcnt), B-fragments reused over 4
//    row-tiles per wave; two independent accumulator chains interleaved to
//    hide WMMA RAW latency.
//  - Self-loop + bias fused into the GEMM epilogue.
//  - Graph aggregation: one wave per edge, float4 gather, f32 atomics
//    (h and accumulator fit in the 192MB L2).
// ---------------------------------------------------------------------------

typedef __attribute__((ext_vector_type(16))) __bf16 v16bf;
typedef __attribute__((ext_vector_type(8)))  __bf16 v8bf;
typedef __attribute__((ext_vector_type(8)))  float  v8f;

#define D 128   // feature dim (D_IN == D_H == D_OUT == 128)

// ---- degree / normalization ------------------------------------------------
__global__ void k_deg_init(float* __restrict__ deg, int N) {
  int i = blockIdx.x * blockDim.x + threadIdx.x;
  if (i < N) deg[i] = 1.0f;                       // self loop contributes 1
}

__global__ void k_deg_accum(const int* __restrict__ dst, float* __restrict__ deg, int E) {
  int e = blockIdx.x * blockDim.x + threadIdx.x;
  if (e < E) atomicAdd(&deg[dst[e]], 1.0f);
}

__global__ void k_dinv(const float* __restrict__ deg, float* __restrict__ dinv, int N) {
  int i = blockIdx.x * blockDim.x + threadIdx.x;
  if (i < N) dinv[i] = rsqrtf(deg[i]);            // deg >= 1 always
}

// ---- precision conversion --------------------------------------------------
__global__ void k_cvt_bf16(const float* __restrict__ in, __bf16* __restrict__ out, int n) {
  int i = blockIdx.x * blockDim.x + threadIdx.x;
  if (i < n) out[i] = (__bf16)in[i];
}

__global__ void k_relu_cvt(const float* __restrict__ in, __bf16* __restrict__ out, int n) {
  int i = blockIdx.x * blockDim.x + threadIdx.x;
  if (i < n) {
    float v = in[i];
    out[i] = (__bf16)(v > 0.0f ? v : 0.0f);
  }
}

// WT[n][k] = (bf16) W[k][n]  -- transposed so B-fragment loads are contiguous
__global__ void k_cvt_wT(const float* __restrict__ W, __bf16* __restrict__ WT) {
  int i = blockIdx.x * blockDim.x + threadIdx.x;  // D*D threads
  int n = i >> 7, k = i & (D - 1);
  WT[n * D + k] = (__bf16)W[k * D + n];
}

// ---- fused WMMA GEMM + self-loop/bias epilogue -----------------------------
// Block = 256 threads = 8 waves; block covers 64 rows x 128 cols.
// Wave w owns column tile [16w, 16w+16); iterates 4 row-tiles (as 2 pairs
// with independent accumulators) reusing its B-fragments held in VGPRs.
//   H[r][c]   = (A @ W)[r][c]                (raw, for edge gather)
//   AGG[r][c] = H[r][c]*dinv[r]^2 + bias[c]  (self-loop + bias init)
__global__ __launch_bounds__(256) void k_gemm_fused(
    const __bf16* __restrict__ A, const __bf16* __restrict__ WT,
    const float* __restrict__ dinv, const float* __restrict__ bias,
    float* __restrict__ H, float* __restrict__ AGG, int N) {
  __shared__ __attribute__((aligned(128))) __bf16 sW[D * D];  // 32 KB staged W^T
  __shared__ float sDinv[64];

  const int tid  = threadIdx.x;
  const int lane = tid & 31;
  const int wave = tid >> 5;
  const int lo   = lane & 15;               // row (A) / column (B, C/D)
  const int hi   = lane >> 4;               // lane-half: K / M group select
  const int row0 = blockIdx.x * 64;

  // ---- async-stage the full 32KB bf16 W^T into LDS (CDNA5 async path).
  // LDS offset operand is derived from &sW[...] via ptrtoint: low 32 bits of
  // the flat LDS address are the LDS byte offset (ISA 10.2), and the ptrtoint
  // captures sW so the compiler must assume the asm writes it.
  {
#pragma unroll
    for (int k = 0; k < 8; ++k) {
      uint32_t ldso = (uint32_t)(uintptr_t)(const void*)(sW + tid * 8 + k * 2048);
      const char* g = (const char*)WT + (size_t)tid * 16u + (size_t)k * 4096;
      asm volatile("global_load_async_to_lds_b128 %0, %1, off"
                   :: "v"(ldso), "v"(g)
                   : "memory");
    }
    asm volatile("s_wait_asynccnt 0x0" ::: "memory");
  }
  if (tid < 64) {                            // dinv for this block's 64 rows
    int r = row0 + tid;
    if (r > N - 1) r = N - 1;
    sDinv[tid] = dinv[r];
  }
  __syncthreads();

  const int col = (wave << 4) + lo;
  const float bcol = bias[col];

  // B 32x16 bf16 fragments (one per K-step), read once from LDS, kept in VGPRs.
  // Lane = column N=lo; K = k0 + hi*16 .. +15 contiguous in transposed W.
  const __bf16* bp = sW + col * D + hi * 16;
  v16bf bfr[4];
  bfr[0] = *(const v16bf*)(bp + 0);
  bfr[1] = *(const v16bf*)(bp + 32);
  bfr[2] = *(const v16bf*)(bp + 64);
  bfr[3] = *(const v16bf*)(bp + 96);

  const bool full = (row0 + 64) <= N;        // uniform: fast store path?

#pragma unroll
  for (int mp = 0; mp < 2; ++mp) {           // two row-tile pairs
    // two independent accumulator chains -> interleaved WMMAs hide RAW latency
    int rowA = row0 + (mp * 2 + 0) * 16 + lo;
    int rowB = row0 + (mp * 2 + 1) * 16 + lo;
    if (rowA > N - 1) rowA = N - 1;          // clamp loads: WMMA needs full EXEC
    if (rowB > N - 1) rowB = N - 1;
    const __bf16* __restrict__ apA = A + (size_t)rowA * D + hi * 16;
    const __bf16* __restrict__ apB = A + (size_t)rowB * D + hi * 16;

    v8f accA = {}, accB = {};
#pragma unroll
    for (int kt = 0; kt < 4; ++kt) {
      // A 16x32 bf16 fragment: lanes 0-15 K={k0..k0+15},
      // lanes 16-31 K={k0+16..k0+31} -> two 16B chunks.
      v8bf a0 = *(const v8bf*)(apA + kt * 32);
      v8bf a1 = *(const v8bf*)(apA + kt * 32 + 8);
      v8bf b0 = *(const v8bf*)(apB + kt * 32);
      v8bf b1 = *(const v8bf*)(apB + kt * 32 + 8);
      v16bf fa, fb;
#pragma unroll
      for (int i = 0; i < 8; ++i) {
        fa[i] = a0[i]; fa[i + 8] = a1[i];
        fb[i] = b0[i]; fb[i + 8] = b1[i];
      }
      accA = __builtin_amdgcn_wmma_f32_16x16x32_bf16(
          false, fa, false, bfr[kt], (short)0, accA, false, false);
      accB = __builtin_amdgcn_wmma_f32_16x16x32_bf16(
          false, fb, false, bfr[kt], (short)0, accB, false, false);
    }

    // C/D layout: lane lo = column, VGPR v = row (hi*8 + v)
#pragma unroll
    for (int half = 0; half < 2; ++half) {
      const v8f& acc = half ? accB : accA;
      const int rbase = row0 + (mp * 2 + half) * 16 + hi * 8;
      if (full) {
#pragma unroll
        for (int v = 0; v < 8; ++v) {
          const int r = rbase + v;
          const float hv = acc[v];
          const float w = sDinv[r - row0];
          H[(size_t)r * D + col]   = hv;
          AGG[(size_t)r * D + col] = hv * (w * w) + bcol;
        }
      } else {
#pragma unroll
        for (int v = 0; v < 8; ++v) {
          const int r = rbase + v;
          if (r < N) {
            const float hv = acc[v];
            const float w = sDinv[r - row0];
            H[(size_t)r * D + col]   = hv;
            AGG[(size_t)r * D + col] = hv * (w * w) + bcol;
          }
        }
      }
    }
  }
}

// ---- edge aggregation ------------------------------------------------------
// One wave per edge: float4 gather (512B/wave) + 4 f32 atomics per lane.
// src/dst/dinv loads are wave-uniform -> scalarized by the compiler.
__global__ __launch_bounds__(256) void k_edge_scatter(
    const float* __restrict__ h, const float* __restrict__ dinv,
    const int* __restrict__ src, const int* __restrict__ dst,
    float* __restrict__ out, int E) {
  int e = blockIdx.x * 8 + (threadIdx.x >> 5);
  if (e >= E) return;
  const int lane = threadIdx.x & 31;
  const int s = src[e], t = dst[e];
  const float w = dinv[s] * dinv[t];
  const float4 v = ((const float4*)(h + (size_t)s * D))[lane];
  float* op = out + (size_t)t * D + lane * 4;
  atomicAdd(op + 0, v.x * w);
  atomicAdd(op + 1, v.y * w);
  atomicAdd(op + 2, v.z * w);
  atomicAdd(op + 3, v.w * w);
}

// ---------------------------------------------------------------------------
extern "C" void kernel_launch(void* const* d_in, const int* in_sizes, int n_in,
                              void* d_out, int out_size, void* d_ws, size_t ws_size,
                              hipStream_t stream) {
  const float* x  = (const float*)d_in[0];
  const float* W1 = (const float*)d_in[1];
  const float* b1 = (const float*)d_in[2];
  const float* W2 = (const float*)d_in[3];
  const float* b2 = (const float*)d_in[4];
  const int*   ei = (const int*)d_in[5];        // edge_index [2, E]

  const int N  = in_sizes[0] / D;               // 100000
  const int E  = in_sizes[5] / 2;               // 600000
  const int ND = N * D;
  const int* src = ei;
  const int* dst = ei + E;
  float* outp = (float*)d_out;

  // workspace carve (256B aligned regions), total ~129 MB
  size_t off = 0;
  auto carve = [&](size_t bytes) -> void* {
    void* p = (char*)d_ws + off;
    off += (bytes + 255) & ~(size_t)255;
    return p;
  };
  float*  deg  = (float*)carve((size_t)N * 4);
  float*  dinv = (float*)carve((size_t)N * 4);
  float*  tmp0 = (float*)carve((size_t)ND * 4);   // raw h = A*W (edge gather src)
  float*  tmp1 = (float*)carve((size_t)ND * 4);   // aggregated layer-1 output
  __bf16* bfA  = (__bf16*)carve((size_t)ND * 2);  // bf16 GEMM input
  __bf16* W1T  = (__bf16*)carve((size_t)D * D * 2);
  __bf16* W2T  = (__bf16*)carve((size_t)D * D * 2);
  (void)ws_size; (void)n_in; (void)out_size;

  const int TB  = 256;
  const int gN  = (N  + TB - 1) / TB;
  const int gE  = (E  + TB - 1) / TB;
  const int gND = (ND + TB - 1) / TB;
  const int gW  = (D * D + TB - 1) / TB;
  const int gGemm = (N + 63) / 64;
  const int gScat = (E + 7) / 8;

  // normalization
  k_deg_init <<<gN, TB, 0, stream>>>(deg, N);
  k_deg_accum<<<gE, TB, 0, stream>>>(dst, deg, E);
  k_dinv     <<<gN, TB, 0, stream>>>(deg, dinv, N);

  // precision prep
  k_cvt_bf16<<<gND, TB, 0, stream>>>(x, bfA, ND);
  k_cvt_wT  <<<gW,  TB, 0, stream>>>(W1, W1T);
  k_cvt_wT  <<<gW,  TB, 0, stream>>>(W2, W2T);

  // layer 1: fused GEMM(+self/bias) -> edge scatter -> ReLU+cvt
  k_gemm_fused  <<<gGemm, 256, 0, stream>>>(bfA, W1T, dinv, b1, tmp0, tmp1, N);
  k_edge_scatter<<<gScat, 256, 0, stream>>>(tmp0, dinv, src, dst, tmp1, E);
  k_relu_cvt    <<<gND, TB, 0, stream>>>(tmp1, bfA, ND);

  // layer 2: fused GEMM(+self/bias into d_out) -> edge scatter into d_out
  k_gemm_fused  <<<gGemm, 256, 0, stream>>>(bfA, W2T, dinv, b2, tmp0, outp, N);
  k_edge_scatter<<<gScat, 256, 0, stream>>>(tmp0, dinv, src, dst, outp, E);
}
